// CrossAttention_27573690040724
// MI455X (gfx1250) — matrix operand order repaired and checked by reading, not compiled, not run on hardware
//
#include <hip/hip_runtime.h>
#include <hip/hip_bf16.h>

typedef unsigned short u16;
typedef unsigned int   u32;
typedef unsigned long long u64;
typedef __attribute__((ext_vector_type(16))) __bf16 v16bf;
typedef __attribute__((ext_vector_type(8)))  float  v8f;
typedef __attribute__((ext_vector_type(4)))  u32    sgx4;
typedef __attribute__((ext_vector_type(8)))  u32    sgx8;

union Frag { v16bf v; uint4 q[2]; };

__device__ __forceinline__ u16 f2bf(float x) {
  union { __bf16 h; u16 u; } c;
  c.h = (__bf16)x;                 // native v_cvt_*bf16_f32 on gfx1250
  return c.u;
}

__device__ __forceinline__ v8f vzero() {
  v8f z;
#pragma unroll
  for (int i = 0; i < 8; ++i) z[i] = 0.0f;
  return z;
}

__device__ __forceinline__ v8f wmma_bf16(const Frag& a, const Frag& b, v8f c) {
  return __builtin_amdgcn_wmma_f32_16x16x32_bf16(false, a.v, false, b.v,
                                                 (short)0, c, false, false);
}

__device__ __forceinline__ uint4 pack8(const float* v) {
  uint4 r;
  r.x = (u32)f2bf(v[0]) | ((u32)f2bf(v[1]) << 16);
  r.y = (u32)f2bf(v[2]) | ((u32)f2bf(v[3]) << 16);
  r.z = (u32)f2bf(v[4]) | ((u32)f2bf(v[5]) << 16);
  r.w = (u32)f2bf(v[6]) | ((u32)f2bf(v[7]) << 16);
  return r;
}

// ---------------------------------------------------------------------------
// Tensor Data Mover: 2D tile (16-bit elements) global -> LDS.
// D# per CDNA5 ISA ch.8: group0 {count,lds_addr,global_addr,type=2},
// group1 {data_size=2B, tensor_dim0/1, tile_dim0/1, tensor_dim0_stride}.
// 2-group form (VADDR2/3 = NULL) is valid for tensors up to 2D.
// Issue once per workgroup (wave 0); completion tracked by TENSORcnt.
// ---------------------------------------------------------------------------
__device__ __forceinline__ void tdm_load_2d(u32 lds_off, const void* gptr,
                                            u32 tile0, u32 tile1,
                                            u32 ten0, u32 ten1, u64 stride0) {
  const u64 ga = (u64)(size_t)gptr;
  sgx4 g0;
  g0[0] = 1u;                                            // count=1, user mode
  g0[1] = lds_off;                                       // lds_addr (bytes)
  g0[2] = (u32)ga;                                       // global_addr[31:0]
  g0[3] = ((u32)(ga >> 32) & 0x01FFFFFFu) | (2u << 30);  // addr[56:32] | type=2
  sgx8 g1;
  g1[0] = 1u << 16;                                      // data_size = 2 bytes
  g1[1] = (ten0 & 0xFFFFu) << 16;                        // tensor_dim0[15:0]
  g1[2] = ((ten0 >> 16) & 0xFFFFu) | ((ten1 & 0xFFFFu) << 16);
  g1[3] = ((ten1 >> 16) & 0xFFFFu) | ((tile0 & 0xFFFFu) << 16);  // tile_dim0
  g1[4] = tile1 & 0xFFFFu;                               // tile_dim1 (tile_dim2=0)
  g1[5] = (u32)stride0;                                  // dim0_stride[31:0]
  g1[6] = (u32)(stride0 >> 32) & 0xFFFFu;                // dim0_stride[47:32]
  g1[7] = 0u;
  asm volatile("tensor_load_to_lds %0, %1" :: "s"(g0), "s"(g1) : "memory");
}

// ---------------------------------------------------------------------------
// Generic GEMM, Kdim = Nout = 1024.  C = A @ B, A: [M][1024], B: [1024][1024].
// ABF16: A stored bf16 (staged via TDM); else f32, converted on stage.
// OMODE 0: bf16 head-major  out[(col>>6)*M*64 + row*64 + (col&63)]
// OMODE 1: bf16 head-transposed out[(col>>6)*64*M + (col&63)*M + row]
// OMODE 2: f32 row-major out[row*1024 + col]
// Block: 256 threads = 8 waves; tile 128(M) x 64(N), K-step 32.
// ---------------------------------------------------------------------------
template<bool ABF16, int OMODE>
__global__ __launch_bounds__(256) void gemm1024(const void* __restrict__ Ap,
                                                const float* __restrict__ B,
                                                void* __restrict__ Cp,
                                                int M, float scale) {
  __shared__ __attribute__((aligned(128))) u16 lA[128][32];
  __shared__ __attribute__((aligned(128))) u16 lB[64][32];   // B transposed: [n][k]

  const int t      = threadIdx.x;
  const int wave   = t >> 5;
  const int lane   = t & 31;
  const int lane15 = lane & 15;
  const int halfl  = lane >> 4;
  const int m0     = blockIdx.x * 128;
  const int n0     = blockIdx.y * 64;

  v8f acc[4];
#pragma unroll
  for (int i = 0; i < 4; ++i) acc[i] = vzero();

  const int arow = t >> 1, ahalf = t & 1;  // A staging (f32 path)
  const int bn   = t & 63, bkg  = t >> 6;  // B staging: 64 cols x 4 groups of 8 k
  const u32 lAoff = (u32)(size_t)&lA[0][0];

  for (int k0 = 0; k0 < 1024; k0 += 32) {
    __syncthreads();
    // ---- stage A tile ----
    if (ABF16) {
      if (wave == 0)
        tdm_load_2d(lAoff, (const u16*)Ap + (size_t)m0 * 1024 + k0,
                    32, 128, 32, 128, 1024);
    } else {
      const float* src = (const float*)Ap + (size_t)(m0 + arow) * 1024 + k0 + ahalf * 16;
      float tmp[16];
#pragma unroll
      for (int i = 0; i < 16; i += 4) {
        float4 f = ((const float4*)src)[i >> 2];
        tmp[i] = f.x; tmp[i + 1] = f.y; tmp[i + 2] = f.z; tmp[i + 3] = f.w;
      }
      *(uint4*)&lA[arow][ahalf * 16 + 0] = pack8(tmp);
      *(uint4*)&lA[arow][ahalf * 16 + 8] = pack8(tmp + 8);
    }
    // ---- stage B tile transposed (f32 -> bf16) ----
    {
      float tmp[8];
#pragma unroll
      for (int i = 0; i < 8; ++i)
        tmp[i] = B[(size_t)(k0 + bkg * 8 + i) * 1024 + n0 + bn];
      *(uint4*)&lB[bn][bkg * 8] = pack8(tmp);
    }
    if (!ABF16 && k0 + 32 < 1024)
      __builtin_prefetch((const float*)Ap + (size_t)(m0 + arow) * 1024 + k0 + 32, 0, 1);
    if (ABF16) __builtin_amdgcn_s_wait_tensorcnt(0);
    __syncthreads();

    // ---- compute: wave owns m-tile = wave*16, all 4 n-tiles ----
    Frag a;
    a.q[0] = *(const uint4*)&lA[wave * 16 + lane15][8 * halfl];
    a.q[1] = *(const uint4*)&lA[wave * 16 + lane15][16 + 8 * halfl];
#pragma unroll
    for (int tn = 0; tn < 4; ++tn) {
      Frag b;
      b.q[0] = *(const uint4*)&lB[tn * 16 + lane15][8 * halfl];
      b.q[1] = *(const uint4*)&lB[tn * 16 + lane15][16 + 8 * halfl];
      acc[tn] = wmma_bf16(a, b, acc[tn]);
    }
  }

  // ---- epilogue ----
#pragma unroll
  for (int tn = 0; tn < 4; ++tn) {
    const int col = n0 + tn * 16 + lane15;
    float v[8];
#pragma unroll
    for (int r = 0; r < 8; ++r) v[r] = acc[tn][r] * scale;

    if (OMODE == 2) {
      float* C = (float*)Cp;
#pragma unroll
      for (int r = 0; r < 8; ++r) {
        const int row = m0 + wave * 16 + 8 * halfl + r;
        C[(size_t)row * 1024 + col] = v[r];
      }
    } else if (OMODE == 0) {
      u16* C = (u16*)Cp;
      const int head = col >> 6, d = col & 63;
#pragma unroll
      for (int r = 0; r < 8; ++r) {
        const int row = m0 + wave * 16 + 8 * halfl + r;
        C[((size_t)head * M + row) * 64 + d] = f2bf(v[r]);
      }
    } else {
      u16* C = (u16*)Cp;
      const int head = col >> 6, d = col & 63;
      const int rowb = m0 + wave * 16 + 8 * halfl;
      *(uint4*)(C + ((size_t)head * 64 + d) * M + rowb) = pack8(v);
    }
  }
}

// ---------------------------------------------------------------------------
// Flash cross-attention for one direction.
// Q : [H][NQ][64] bf16 (pre-scaled by 1/sqrt(DK))
// K : [H][NK][64] bf16
// VT: [H][64][NK] bf16 (V transposed per head)
// CT: [H*64][NQ]  bf16 (context transposed == torch-faithful scrambled layout)
// Block: 256 threads = 8 waves; each wave owns 16 queries; key chunks of 64.
// K/V tiles staged by the Tensor Data Mover, double-buffered in LDS.
// ---------------------------------------------------------------------------
__global__ __launch_bounds__(256) void attn_fwd(const u16* __restrict__ Q,
                                                const u16* __restrict__ K,
                                                const u16* __restrict__ VT,
                                                u16* __restrict__ CT,
                                                int NQ, int NK) {
  __shared__ __attribute__((aligned(128))) u16 lK[2][64][64];   // keys x dk
  __shared__ __attribute__((aligned(128))) u16 lV[2][64][64];   // d x keys
  __shared__ __attribute__((aligned(128))) u16 lP[8][16][64];   // per-wave probs

  const int t      = threadIdx.x;
  const int wave   = t >> 5;
  const int lane   = t & 31;
  const int lane15 = lane & 15;
  const int halfl  = lane >> 4;
  const int head   = blockIdx.y;
  const int q0     = blockIdx.x * 128 + wave * 16;

  const u16* Kbase = K  + (size_t)head * NK * 64;
  const u16* Vbase = VT + (size_t)head * 64 * NK;
  const u32 lkOff0 = (u32)(size_t)&lK[0][0][0];
  const u32 lkOff1 = (u32)(size_t)&lK[1][0][0];
  const u32 lvOff0 = (u32)(size_t)&lV[0][0][0];
  const u32 lvOff1 = (u32)(size_t)&lV[1][0][0];
  const int nch = NK >> 6;

  // kick off TDM for chunk 0 while Q fragments load
  if (wave == 0) {
    tdm_load_2d(lkOff0, Kbase, 64, 64, 64, 64, 64);
    tdm_load_2d(lvOff0, Vbase, 64, 64, 64, 64, (u64)NK);
  }

  // Q A-fragments for this wave's 16 queries (kept in registers whole kernel)
  Frag qa[2];
  {
    const u16* qr = Q + ((size_t)head * NQ + q0 + lane15) * 64;
#pragma unroll
    for (int s = 0; s < 2; ++s) {
      qa[s].q[0] = *(const uint4*)(qr + 32 * s + 8 * halfl);
      qa[s].q[1] = *(const uint4*)(qr + 32 * s + 16 + 8 * halfl);
    }
  }

  float mrun[8], lrun[8];
#pragma unroll
  for (int r = 0; r < 8; ++r) { mrun[r] = -__builtin_inff(); lrun[r] = 0.0f; }
  v8f acc[4];
#pragma unroll
  for (int i = 0; i < 4; ++i) acc[i] = vzero();

  __builtin_amdgcn_s_wait_tensorcnt(0);
  __syncthreads();

  int buf = 0;
  for (int c = 0; c < nch; ++c) {
    // prefetch next chunk via TDM into the other buffer (overlaps compute)
    if (wave == 0 && c + 1 < nch) {
      const u32 lk = buf ? lkOff0 : lkOff1;
      const u32 lv = buf ? lvOff0 : lvOff1;
      tdm_load_2d(lk, Kbase + (size_t)(c + 1) * 64 * 64, 64, 64, 64, 64, 64);
      tdm_load_2d(lv, Vbase + (size_t)(c + 1) * 64,      64, 64, 64, 64, (u64)NK);
    }
    const u16 (*cK)[64] = lK[buf];
    const u16 (*cV)[64] = lV[buf];

    // ---- S = Q K^T for four 16-key tiles (dk contraction = 2 wmma steps) ----
    v8f s[4];
#pragma unroll
    for (int j = 0; j < 4; ++j) {
      Frag b0, b1;
      const int key = j * 16 + lane15;
      b0.q[0] = *(const uint4*)&cK[key][8 * halfl];
      b0.q[1] = *(const uint4*)&cK[key][16 + 8 * halfl];
      b1.q[0] = *(const uint4*)&cK[key][32 + 8 * halfl];
      b1.q[1] = *(const uint4*)&cK[key][48 + 8 * halfl];
      v8f sj = vzero();
      sj = wmma_bf16(qa[0], b0, sj);
      sj = wmma_bf16(qa[1], b1, sj);
      s[j] = sj;
    }

    // ---- online softmax (row reductions across half-wave lanes) ----
    float mnew[8], alpha[8], lsum[8];
#pragma unroll
    for (int r = 0; r < 8; ++r) {
      float v = fmaxf(fmaxf(s[0][r], s[1][r]), fmaxf(s[2][r], s[3][r]));
      v = fmaxf(v, __shfl_xor(v, 1));
      v = fmaxf(v, __shfl_xor(v, 2));
      v = fmaxf(v, __shfl_xor(v, 4));
      v = fmaxf(v, __shfl_xor(v, 8));
      mnew[r]  = fmaxf(mrun[r], v);
      alpha[r] = __expf(mrun[r] - mnew[r]);
      mrun[r]  = mnew[r];
      lsum[r]  = 0.0f;
    }
#pragma unroll
    for (int j = 0; j < 4; ++j) {
#pragma unroll
      for (int r = 0; r < 8; ++r) {
        const float p = __expf(s[j][r] - mnew[r]);
        lsum[r] += p;
        lP[wave][8 * halfl + r][j * 16 + lane15] = f2bf(p);
      }
    }
#pragma unroll
    for (int r = 0; r < 8; ++r) {
      float v = lsum[r];
      v += __shfl_xor(v, 1);
      v += __shfl_xor(v, 2);
      v += __shfl_xor(v, 4);
      v += __shfl_xor(v, 8);
      lrun[r] = lrun[r] * alpha[r] + v;
#pragma unroll
      for (int tn = 0; tn < 4; ++tn) acc[tn][r] *= alpha[r];
    }

    // ---- context += P @ V (keys contraction = 2 wmma steps of 32) ----
    Frag pa[2];
#pragma unroll
    for (int sx = 0; sx < 2; ++sx) {
      pa[sx].q[0] = *(const uint4*)&lP[wave][lane15][32 * sx + 8 * halfl];
      pa[sx].q[1] = *(const uint4*)&lP[wave][lane15][32 * sx + 16 + 8 * halfl];
    }
#pragma unroll
    for (int tn = 0; tn < 4; ++tn) {
      const int d = tn * 16 + lane15;
      Frag b0, b1;
      b0.q[0] = *(const uint4*)&cV[d][8 * halfl];
      b0.q[1] = *(const uint4*)&cV[d][16 + 8 * halfl];
      b1.q[0] = *(const uint4*)&cV[d][32 + 8 * halfl];
      b1.q[1] = *(const uint4*)&cV[d][48 + 8 * halfl];
      acc[tn] = wmma_bf16(pa[0], b0, acc[tn]);
      acc[tn] = wmma_bf16(pa[1], b1, acc[tn]);
    }

    __builtin_amdgcn_s_wait_tensorcnt(0);  // wave0's prefetch done before swap
    __syncthreads();
    buf ^= 1;
  }

  // ---- normalize and store transposed context: CT[(h*64+d)][q] ----
#pragma unroll
  for (int tn = 0; tn < 4; ++tn) {
    const int d = tn * 16 + lane15;
    float v[8];
#pragma unroll
    for (int r = 0; r < 8; ++r) v[r] = acc[tn][r] / lrun[r];
    *(uint4*)(CT + ((size_t)head * 64 + d) * NQ + q0 + 8 * halfl) = pack8(v);
  }
}

// ---------------------------------------------------------------------------
extern "C" void kernel_launch(void* const* d_in, const int* in_sizes, int n_in,
                              void* d_out, int out_size, void* d_ws, size_t ws_size,
                              hipStream_t stream) {
  (void)in_sizes; (void)n_in; (void)out_size; (void)ws_size;

  const float* X    = (const float*)d_in[0];
  const float* X1   = (const float*)d_in[1];
  const float* WQ   = (const float*)d_in[2];
  const float* WK   = (const float*)d_in[3];
  const float* WV   = (const float*)d_in[4];
  const float* WQ1  = (const float*)d_in[5];
  const float* WK1  = (const float*)d_in[6];
  const float* WV1  = (const float*)d_in[7];
  const float* Wfc  = (const float*)d_in[8];
  const float* Wfc1 = (const float*)d_in[9];
  float* out = (float*)d_out;

  const int N0 = 2048, N1 = 3072;
  u16* Qb   = (u16*)d_ws;
  u16* Kb   = Qb   + (size_t)16 * N0 * 64;
  u16* VTb  = Kb   + (size_t)16 * N0 * 64;
  u16* Q1b  = VTb  + (size_t)16 * N0 * 64;
  u16* K1b  = Q1b  + (size_t)16 * N1 * 64;
  u16* V1Tb = K1b  + (size_t)16 * N1 * 64;
  u16* CT1  = V1Tb + (size_t)16 * N1 * 64;
  u16* CT2  = CT1  + (size_t)1024 * N1;

  dim3 blk(256);
  const float qs = 0.125f;  // 1/sqrt(DK) folded into Q projection

  gemm1024<false, 0><<<dim3(N0 / 128, 16), blk, 0, stream>>>(X,  WQ,  Qb,   N0, qs);
  gemm1024<false, 0><<<dim3(N0 / 128, 16), blk, 0, stream>>>(X,  WK,  Kb,   N0, 1.0f);
  gemm1024<false, 1><<<dim3(N0 / 128, 16), blk, 0, stream>>>(X,  WV,  VTb,  N0, 1.0f);
  gemm1024<false, 0><<<dim3(N1 / 128, 16), blk, 0, stream>>>(X1, WQ1, Q1b,  N1, qs);
  gemm1024<false, 0><<<dim3(N1 / 128, 16), blk, 0, stream>>>(X1, WK1, K1b,  N1, 1.0f);
  gemm1024<false, 1><<<dim3(N1 / 128, 16), blk, 0, stream>>>(X1, WV1, V1Tb, N1, 1.0f);

  attn_fwd<<<dim3(N1 / 128, 16), blk, 0, stream>>>(Q1b, Kb,  VTb,  CT1, N1, N0);
  attn_fwd<<<dim3(N0 / 128, 16), blk, 0, stream>>>(Qb,  K1b, V1Tb, CT2, N0, N1);

  gemm1024<true, 2><<<dim3(N1 / 128, 16), blk, 0, stream>>>(CT1, Wfc,  out,                     N1, 1.0f);
  gemm1024<true, 2><<<dim3(N0 / 128, 16), blk, 0, stream>>>(CT2, Wfc1, out + (size_t)N1 * 1024, N0, 1.0f);
}
